// vgae_23433341567203
// MI455X (gfx1250) — matrix operand order repaired and checked by reading, not compiled, not run on hardware
//
#include <hip/hip_runtime.h>
#include <cstdint>

#define D 128
#define EDGES 640000
#define TILE_M 16
#define WAVES_PER_BLOCK 4
#define TILES_PER_WAVE 8
#define TILES_PER_BLOCK (WAVES_PER_BLOCK * TILES_PER_WAVE)   // 32
#define N_TILES (EDGES / TILE_M)                             // 40000
#define POS_BLOCKS (N_TILES / TILES_PER_BLOCK)               // 1250
#define WROW 68   // dwords per LDS weight row: 136 bf16 (128 valid + 8 pad), 272B keeps 16B align

#define USE_ASYNC_LDS 1   // stage weights with global_load_async_to_lds_b128

typedef __attribute__((ext_vector_type(16))) __bf16 v16bf;
typedef __attribute__((ext_vector_type(8)))  float  v8f;

union BFrag { uint32_t u[8]; v16bf v; };
union AFrag { __bf16 h[16]; v16bf v; };

__device__ __forceinline__ float sigmoidf(float z) {
  return 1.0f / (1.0f + __expf(-z));
}

// Convert W1 / We1 (f32, [k][n]) -> bf16 transposed [n][k] in workspace.
__global__ void convert_weights_kernel(const float* __restrict__ W1,
                                       const float* __restrict__ We1,
                                       uint16_t* __restrict__ wbf) {
  int tid = blockIdx.x * blockDim.x + threadIdx.x;
  if (tid >= 2 * D * D) return;
  int mat   = tid >> 14;           // 0 = W1, 1 = We1
  int local = tid & (D * D - 1);
  int k = local >> 7;
  int n = local & (D - 1);
  const float* src = mat ? We1 : W1;
  __bf16 b = (__bf16)src[k * D + n];          // native cvt (RNE)
  union { __bf16 h; uint16_t u; } c; c.h = b;
  wbf[mat * D * D + n * D + k] = c.u;
}

__global__ __launch_bounds__(WAVES_PER_BLOCK * 32)
void vgae_main_kernel(const float* __restrict__ x,
                      const int*   __restrict__ ei,
                      const int*   __restrict__ ein,
                      const float* __restrict__ b1,
                      const float* __restrict__ W2,
                      const float* __restrict__ b2,
                      const float* __restrict__ be1,
                      const float* __restrict__ We2,
                      const float* __restrict__ be2,
                      const uint16_t* __restrict__ wbf,   // [W1T | We1T] bf16 [n][k]
                      float* __restrict__ out_attr,       // [EDGES][4]
                      float* __restrict__ out_pos,        // [EDGES]
                      float* __restrict__ out_neg)        // [EDGES]
{
  __shared__ uint32_t sW1T [D * WROW];   // bf16 pairs, [n][k], padded rows (16B-aligned rows)
  __shared__ uint32_t sWe1T[D * WROW];
  __shared__ float sb1[D], sbe1[D], sWe2[D];
  __shared__ __align__(16) float sW2[D * 4];

  const bool neg = (blockIdx.x >= POS_BLOCKS);
  const int tid  = threadIdx.x;

  // ---- stage weights into LDS ----
#if USE_ASYNC_LDS
  {
    // 16B chunks: per matrix 128 rows x 16 chunks = 2048
    const int nch = neg ? 2048 : 4096;
    for (int i = tid; i < nch; i += blockDim.x) {
      const int j      = neg ? (i + 2048) : i;  // chunk id in [W1T | We1T] space
      const int mat    = j >> 11;               // 0 = W1T, 1 = We1T
      const int within = j & 2047;
      const int n  = within >> 4;
      const int kp = within & 15;
      const uint16_t* g = wbf + mat * D * D + n * D + kp * 8;
      uint32_t* sdst = mat ? sWe1T : sW1T;
      uint32_t laddr = (uint32_t)(uintptr_t)(&sdst[n * WROW + kp * 4]); // LDS offset = low 32b
      uint64_t gaddr = (uint64_t)(uintptr_t)g;
      asm volatile("global_load_async_to_lds_b128 %0, %1, off"
                   :: "v"(laddr), "v"(gaddr) : "memory");
    }
    asm volatile("s_wait_asynccnt 0x0" ::: "memory");
  }
#else
  {
    const uint32_t* gWe1 = (const uint32_t*)(wbf + D * D);
    for (int i = tid; i < D * 64; i += blockDim.x) {
      int n = i >> 6, kp = i & 63;
      sWe1T[n * WROW + kp] = gWe1[i];
    }
    if (!neg) {
      const uint32_t* gW1 = (const uint32_t*)wbf;
      for (int i = tid; i < D * 64; i += blockDim.x) {
        int n = i >> 6, kp = i & 63;
        sW1T[n * WROW + kp] = gW1[i];
      }
    }
  }
#endif
  for (int i = tid; i < D; i += blockDim.x) { sbe1[i] = be1[i]; sWe2[i] = We2[i]; }
  if (!neg) {
    for (int i = tid; i < D; i += blockDim.x)     sb1[i] = b1[i];
    for (int i = tid; i < D * 4; i += blockDim.x) sW2[i] = W2[i];
  }
  __syncthreads();

  const int lane = tid & 31;
  const int wave = tid >> 5;
  const int nlo  = lane & 15;   // column within N-tile / row within M-tile
  const int hi   = lane >> 4;   // half-wave selector
  const int blk  = neg ? (blockIdx.x - POS_BLOCKS) : blockIdx.x;
  const int* eidx = neg ? ein : ei;
  float* oedge    = neg ? out_neg : out_pos;

  const float b20 = b2[0], b21 = b2[1], b22 = b2[2], b23 = b2[3];
  const float be2v = be2[0];

#pragma unroll 1
  for (int it = 0; it < TILES_PER_WAVE; ++it) {
    const int tile = blk * TILES_PER_BLOCK + wave * TILES_PER_WAVE + it;
    const int row0 = tile * TILE_M;

    // ---- gather + relu(x_s * x_d) directly into 16x32 bf16 A fragments ----
    const int eg = row0 + nlo;
    const int si = eidx[eg];
    const int di = eidx[EDGES + eg];
    const float* xs = x + (size_t)si * D;
    const float* xd = x + (size_t)di * D;

    v16bf A[4];
#pragma unroll
    for (int c = 0; c < 4; ++c) {
      AFrag f;
      const int kb = 32 * c + 8 * hi;
#pragma unroll
      for (int i = 0; i < 8; ++i) {
        const int k = kb + ((i < 4) ? 2 * i : 16 + 2 * (i - 4));
        float2 a = *(const float2*)(xs + k);
        float2 b = *(const float2*)(xd + k);
        f.h[2 * i]     = (__bf16)fmaxf(a.x * b.x, 0.0f);   // native v_cvt_pk_bf16_f32
        f.h[2 * i + 1] = (__bf16)fmaxf(a.y * b.y, 0.0f);
      }
      A[c] = f.v;
    }

    // ======== edge decoder: relu(m @ We1 + be1) @ We2 + be2 ========
    float acce[8];
#pragma unroll
    for (int r = 0; r < 8; ++r) acce[r] = 0.0f;

#pragma unroll 2
    for (int t = 0; t < 8; ++t) {
      v8f dmat = {};
      const uint32_t* wrow = sWe1T + (16 * t + nlo) * WROW;
#pragma unroll
      for (int c = 0; c < 4; ++c) {
        BFrag bf;
        const int kd = 16 * c + 4 * hi;
#pragma unroll
        for (int v = 0; v < 8; ++v) bf.u[v] = wrow[kd + v + ((v < 4) ? 0 : 4)];
        dmat = __builtin_amdgcn_wmma_f32_16x16x32_bf16(
            false, A[c], false, bf.v, (short)0, dmat, false, false);
      }
      const float bias = sbe1[16 * t + nlo];
      const float w    = sWe2[16 * t + nlo];
#pragma unroll
      for (int r = 0; r < 8; ++r) {
        float h = fmaxf(dmat[r] + bias, 0.0f);
        acce[r] += h * w;
      }
    }
#pragma unroll
    for (int r = 0; r < 8; ++r) {        // reduce over 16 columns
      float v = acce[r];
      v += __shfl_xor(v, 1); v += __shfl_xor(v, 2);
      v += __shfl_xor(v, 4); v += __shfl_xor(v, 8);
      acce[r] = v;
    }
    if (nlo == 0) {
      const int rb = row0 + 8 * hi;
#pragma unroll
      for (int r = 0; r < 8; ++r)
        oedge[rb + r] = sigmoidf(acce[r] + be2v);
    }

    // ======== attr decoder (pos only): sigmoid(relu(m@W1+b1) @ W2 + b2) ========
    if (!neg) {
      float ac0[8], ac1[8], ac2[8], ac3[8];
#pragma unroll
      for (int r = 0; r < 8; ++r) { ac0[r] = ac1[r] = ac2[r] = ac3[r] = 0.0f; }

#pragma unroll 2
      for (int t = 0; t < 8; ++t) {
        v8f dmat = {};
        const uint32_t* wrow = sW1T + (16 * t + nlo) * WROW;
#pragma unroll
        for (int c = 0; c < 4; ++c) {
          BFrag bf;
          const int kd = 16 * c + 4 * hi;
#pragma unroll
          for (int v = 0; v < 8; ++v) bf.u[v] = wrow[kd + v + ((v < 4) ? 0 : 4)];
          dmat = __builtin_amdgcn_wmma_f32_16x16x32_bf16(
              false, A[c], false, bf.v, (short)0, dmat, false, false);
        }
        const float bias = sb1[16 * t + nlo];
        const int   nn   = 16 * t + nlo;
        const float w0 = sW2[nn * 4 + 0], w1 = sW2[nn * 4 + 1];
        const float w2 = sW2[nn * 4 + 2], w3 = sW2[nn * 4 + 3];
#pragma unroll
        for (int r = 0; r < 8; ++r) {
          float h = fmaxf(dmat[r] + bias, 0.0f);
          ac0[r] += h * w0; ac1[r] += h * w1; ac2[r] += h * w2; ac3[r] += h * w3;
        }
      }
#pragma unroll
      for (int r = 0; r < 8; ++r) {
        float v0 = ac0[r], v1 = ac1[r], v2 = ac2[r], v3 = ac3[r];
        v0 += __shfl_xor(v0, 1); v0 += __shfl_xor(v0, 2); v0 += __shfl_xor(v0, 4); v0 += __shfl_xor(v0, 8);
        v1 += __shfl_xor(v1, 1); v1 += __shfl_xor(v1, 2); v1 += __shfl_xor(v1, 4); v1 += __shfl_xor(v1, 8);
        v2 += __shfl_xor(v2, 1); v2 += __shfl_xor(v2, 2); v2 += __shfl_xor(v2, 4); v2 += __shfl_xor(v2, 8);
        v3 += __shfl_xor(v3, 1); v3 += __shfl_xor(v3, 2); v3 += __shfl_xor(v3, 4); v3 += __shfl_xor(v3, 8);
        ac0[r] = v0; ac1[r] = v1; ac2[r] = v2; ac3[r] = v3;
      }
      if (nlo == 0) {
        const int rb = row0 + 8 * hi;
#pragma unroll
        for (int r = 0; r < 8; ++r) {
          float4 o;
          o.x = sigmoidf(ac0[r] + b20);
          o.y = sigmoidf(ac1[r] + b21);
          o.z = sigmoidf(ac2[r] + b22);
          o.w = sigmoidf(ac3[r] + b23);
          *(float4*)(out_attr + (size_t)(rb + r) * 4) = o;
        }
      }
    }
  }
}

extern "C" void kernel_launch(void* const* d_in, const int* in_sizes, int n_in,
                              void* d_out, int out_size, void* d_ws, size_t ws_size,
                              hipStream_t stream) {
  const float* x    = (const float*)d_in[0];
  const int*   ei   = (const int*)d_in[1];
  const int*   ein  = (const int*)d_in[2];
  const float* W1   = (const float*)d_in[3];
  const float* b1   = (const float*)d_in[4];
  const float* W2   = (const float*)d_in[5];
  const float* b2   = (const float*)d_in[6];
  const float* We1  = (const float*)d_in[7];
  const float* be1  = (const float*)d_in[8];
  const float* We2  = (const float*)d_in[9];
  const float* be2  = (const float*)d_in[10];

  uint16_t* wbf = (uint16_t*)d_ws;   // 64 KB: W1T then We1T, bf16 [n][k]

  float* out      = (float*)d_out;
  float* out_attr = out;                        // EDGES*4
  float* out_pos  = out + (size_t)EDGES * 4;    // EDGES
  float* out_neg  = out + (size_t)EDGES * 5;    // EDGES

  convert_weights_kernel<<<(2 * D * D + 255) / 256, 256, 0, stream>>>(W1, We1, wbf);

  vgae_main_kernel<<<2 * POS_BLOCKS, WAVES_PER_BLOCK * 32, 0, stream>>>(
      x, ei, ein, b1, W2, b2, be1, We2, be2, wbf, out_attr, out_pos, out_neg);
}
